// MultiHeadAttentionRelative_30975304138933
// MI455X (gfx1250) — matrix operand order repaired and checked by reading, not compiled, not run on hardware
//
#include <hip/hip_runtime.h>

// Problem constants (match reference)
#define BB 2
#define LL 1024
#define DD 64
#define HH 8
#define MAXD 32
#define NPOS 65
#define QE_LD 80   // 65 padded up to 5x16 tiles
#define AP_LD 96   // attn_pos padded up to 3x32 (WMMA k-dim)
#define CHK 128    // keys per LDS-staged chunk in k_attn
#define NCH (LL / CHK)

typedef __attribute__((ext_vector_type(16))) __bf16 v16bf;
typedef __attribute__((ext_vector_type(8)))  float  v8f;
typedef __attribute__((ext_vector_type(4)))  unsigned int u32x4;
typedef __attribute__((ext_vector_type(4)))  int    i32x4;
typedef __attribute__((ext_vector_type(8)))  int    i32x8;

#if __has_builtin(__builtin_amdgcn_tensor_load_to_lds) && \
    __has_builtin(__builtin_amdgcn_s_wait_tensorcnt)
#define HAVE_TDM 1
#if __has_include(<hip/amd_detail/amd_gfx1250_TDM.h>)
#define TDM_SIX_ARGS 1   // therock-10.0 headers -> clang-23 -> 6-arg builtin
#endif
#else
#define HAVE_TDM 0
#endif

__device__ __forceinline__ v8f wmma_bf16(v16bf a, v16bf b, v8f c) {
  // D(f32,16x16) = A(bf16,16x32) * B(bf16,32x16) + C
  return __builtin_amdgcn_wmma_f32_16x16x32_bf16(false, a, false, b, (short)0, c,
                                                 false, false);
}

// ---- WMMA operand loaders (wave32, contiguous k = 16*(l/16) + s in both A and B, not the §7.12.2 table order) ----

// A 16x32 bf16 tile from row-major bf16 src (ld elems). Lane l: m=l&15,
// elements 0..15 -> k = kb..kb+15, kb = (l<16)?0:16.
__device__ __forceinline__ v16bf load_a_bf16(const __bf16* src, int ld, int r0, int c0) {
  int l  = threadIdx.x & 31;
  int m  = l & 15;
  int kb = (l & 16) ? 16 : 0;
  const __bf16* p = src + (size_t)(r0 + m) * ld + c0 + kb;
  union { uint4 u[2]; v16bf v; } cv;
  cv.u[0] = *(const uint4*)(p);
  cv.u[1] = *(const uint4*)(p + 8);
  return cv.v;
}

// A 16x32 tile from row-major f32 src, converted to bf16.
__device__ __forceinline__ v16bf load_a_f32(const float* src, int ld, int r0, int c0) {
  int l  = threadIdx.x & 31;
  int m  = l & 15;
  int kb = (l & 16) ? 16 : 0;
  const float* p = src + (size_t)(r0 + m) * ld + c0 + kb;
  float4 x0 = *(const float4*)(p);
  float4 x1 = *(const float4*)(p + 4);
  float4 x2 = *(const float4*)(p + 8);
  float4 x3 = *(const float4*)(p + 12);
  v16bf a;
  a[0]=(__bf16)x0.x; a[1]=(__bf16)x0.y; a[2]=(__bf16)x0.z; a[3]=(__bf16)x0.w;
  a[4]=(__bf16)x1.x; a[5]=(__bf16)x1.y; a[6]=(__bf16)x1.z; a[7]=(__bf16)x1.w;
  a[8]=(__bf16)x2.x; a[9]=(__bf16)x2.y; a[10]=(__bf16)x2.z; a[11]=(__bf16)x2.w;
  a[12]=(__bf16)x3.x; a[13]=(__bf16)x3.y; a[14]=(__bf16)x3.z; a[15]=(__bf16)x3.w;
  return a;
}

// B 32x16 bf16 tile where element (k,n) = src[(n0+n)*ld + c0 + k]
// ("n-major" source = B^T stored row-major). Lane l: n=l&15, k = kb..kb+15, kb=(l<16)?0:16.
__device__ __forceinline__ v16bf load_b_nmajor(const __bf16* src, int ld, int n0, int c0) {
  int l  = threadIdx.x & 31;
  int n  = l & 15;
  int kb = (l & 16) ? 16 : 0;
  const __bf16* p = src + (size_t)(n0 + n) * ld + c0 + kb;
  union { uint4 u[2]; v16bf v; } cv;
  cv.u[0] = *(const uint4*)(p);
  cv.u[1] = *(const uint4*)(p + 8);
  return cv.v;
}

// Same, from f32 source with row guard (rows n >= nmax contribute zero).
__device__ __forceinline__ v16bf load_b_f32_nmajor_g(const float* src, int ld, int n0,
                                                     int c0, int nmax) {
  int l  = threadIdx.x & 31;
  int n  = n0 + (l & 15);
  int kb = (l & 16) ? 16 : 0;
  v16bf b;
  if (n < nmax) {
    const float* p = src + (size_t)n * ld + c0 + kb;
#pragma unroll
    for (int j = 0; j < 16; ++j) b[j] = (__bf16)p[j];
  } else {
#pragma unroll
    for (int j = 0; j < 16; ++j) b[j] = (__bf16)0.0f;
  }
  return b;
}

// B 32x16 tile from k-major f32 source: element (k,n) = src[(c0+k)*ld + n0+n],
// rows k >= kmax contribute zero.
__device__ __forceinline__ v16bf load_b_f32_kmajor_g(const float* src, int ld, int c0,
                                                     int n0, int kmax) {
  int l  = threadIdx.x & 31;
  int n  = n0 + (l & 15);
  int kb = c0 + ((l & 16) ? 16 : 0);
  v16bf b;
#pragma unroll
  for (int j = 0; j < 16; ++j) {
    int k = kb + j;
    b[j] = (k < kmax) ? (__bf16)src[(size_t)k * ld + n] : (__bf16)0.0f;
  }
  return b;
}

// C/D f32 stores: lane l holds rows m = mb..mb+7 (mb = (l<16)?0:8), col n = l&15.
__device__ __forceinline__ void store_c_f32(float* dst, int ld, int r0, int c0,
                                            const v8f& c) {
  int l = threadIdx.x & 31;
  int n = l & 15;
  int mb = (l & 16) ? 8 : 0;
#pragma unroll
  for (int r = 0; r < 8; ++r) dst[(size_t)(r0 + mb + r) * ld + c0 + n] = c[r];
}

__device__ __forceinline__ void store_c_bf16(__bf16* dst, int ld, int r0, int c0,
                                             const v8f& c, float scale) {
  int l = threadIdx.x & 31;
  int n = l & 15;
  int mb = (l & 16) ? 8 : 0;
#pragma unroll
  for (int r = 0; r < 8; ++r)
    dst[(size_t)(r0 + mb + r) * ld + c0 + n] = (__bf16)(c[r] * scale);
}

// ---- Tensor Data Mover: 2D bf16 tile (rows x 64) global -> LDS ----------
// D# per CDNA5 ISA. Tracked by TENSORcnt.
#if HAVE_TDM
__device__ __forceinline__ void tdm_load_k_tile(const __bf16* gsrc, unsigned lds_addr,
                                                int rows) {
  unsigned long long ga = (unsigned long long)(uintptr_t)gsrc;
  u32x4 g0;
  g0[0] = 1u;                                        // count=1, user descriptor
  g0[1] = lds_addr;                                  // LDS byte address
  g0[2] = (unsigned)(ga & 0xffffffffu);              // global_addr[31:0]
  g0[3] = (unsigned)((ga >> 32) & 0x01ffffffu) | (2u << 30);  // [56:32] | type=2
  i32x8 g1;
  g1[0] = 0x00010000;            // workgroup_mask=0, data_size=1 (2 bytes)
  g1[1] = (int)(64u << 16);      // tensor_dim0 = 64   (bits 79:48 -> [31:16])
  g1[2] = (int)((unsigned)LL << 16);  // tensor_dim1 = L (bits 111:80 -> [31:16])
  g1[3] = (int)(64u << 16);      // tile_dim0 = 64     (bits 127:112 -> [31:16])
  g1[4] = rows;                  // tile_dim1 = rows, tile_dim2 = 0
  g1[5] = 64;                    // tensor_dim0_stride = 64 (low 32)
  g1[6] = 0;
  g1[7] = 0;
  i32x4 z4 = {0, 0, 0, 0};
#ifdef TDM_SIX_ARGS
  i32x8 z8 = {0, 0, 0, 0, 0, 0, 0, 0};
  __builtin_amdgcn_tensor_load_to_lds(g0, g1, z4, z4, z8, 0);
#else
  __builtin_amdgcn_tensor_load_to_lds(g0, g1, z4, z4, 0);
#endif
}
#endif

// Stage a CHK x 64 bf16 K chunk into LDS (TDM if available, else cooperative copy).
__device__ __forceinline__ void stage_k(const __bf16* Ko, __bf16* buf, int key0) {
#if HAVE_TDM
  if ((threadIdx.x >> 5) == 0)  // one wave issues; TDM ignores EXEC
    tdm_load_k_tile(Ko + (size_t)key0 * DD, (unsigned)(uintptr_t)buf, CHK);
#else
  const unsigned* s = (const unsigned*)(Ko + (size_t)key0 * DD);
  unsigned* d = (unsigned*)buf;
  for (int i = threadIdx.x; i < CHK * DD / 2; i += (int)blockDim.x) d[i] = s[i];
#endif
}

// Release a staged chunk to all waves. `more` = another chunk is in flight.
__device__ __forceinline__ void wait_k(bool more) {
#if HAVE_TDM
  if ((threadIdx.x >> 5) == 0) {
    if (more) __builtin_amdgcn_s_wait_tensorcnt(1);  // oldest (current) chunk done
    else      __builtin_amdgcn_s_wait_tensorcnt(0);
  }
#endif
  __syncthreads();
}

// ---------------- Kernel 1: per-head Q/K/V projections ----------------
// Q = (q @ Wq)/8 -> bf16 row-major (BH,L,D); K likewise; V stored transposed (BH,D,L).
__global__ __launch_bounds__(256) void k_proj(const float* __restrict__ q,
                                              const float* __restrict__ kv,
                                              const float* __restrict__ Wq,
                                              const float* __restrict__ Wk,
                                              const float* __restrict__ Wv,
                                              __bf16* __restrict__ Qbf,
                                              __bf16* __restrict__ Kbf,
                                              __bf16* __restrict__ Vt) {
  __shared__ __align__(16) __bf16 lwq[DD * DD];  // lw[o*64 + i] = W[h][i][o] (bf16)
  __shared__ __align__(16) __bf16 lwk[DD * DD];
  __shared__ __align__(16) __bf16 lwv[DD * DD];
  int bh = blockIdx.x;
  int b = bh / HH, h = bh % HH;
  for (int idx = threadIdx.x; idx < DD * DD; idx += blockDim.x) {
    int i = idx / DD, o = idx % DD;
    size_t wi = ((size_t)h * DD + i) * DD + o;
    lwq[o * DD + i] = (__bf16)Wq[wi];
    lwk[o * DD + i] = (__bf16)Wk[wi];
    lwv[o * DD + i] = (__bf16)Wv[wi];
  }
  __syncthreads();

  int w  = threadIdx.x >> 5;
  int r0 = blockIdx.y * 128 + w * 16;
  const float* xq = q + (size_t)b * LL * DD;
  const float* xk = kv + (size_t)b * LL * DD;
  v16bf aq0 = load_a_f32(xq, DD, r0, 0);
  v16bf aq1 = load_a_f32(xq, DD, r0, 32);
  v16bf ak0 = load_a_f32(xk, DD, r0, 0);
  v16bf ak1 = load_a_f32(xk, DD, r0, 32);

  __bf16* Qo = Qbf + (size_t)bh * LL * DD;
  __bf16* Ko = Kbf + (size_t)bh * LL * DD;
  __bf16* Vo = Vt + (size_t)bh * DD * LL;
  int lane = threadIdx.x & 31;
  int n = lane & 15, mb = (lane & 16) ? 8 : 0;

#pragma unroll
  for (int n0 = 0; n0 < DD; n0 += 16) {
    v8f cq = {}, ck = {}, cv = {};
    cq = wmma_bf16(aq0, load_b_nmajor(lwq, DD, n0, 0), cq);
    cq = wmma_bf16(aq1, load_b_nmajor(lwq, DD, n0, 32), cq);
    ck = wmma_bf16(ak0, load_b_nmajor(lwk, DD, n0, 0), ck);
    ck = wmma_bf16(ak1, load_b_nmajor(lwk, DD, n0, 32), ck);
    cv = wmma_bf16(ak0, load_b_nmajor(lwv, DD, n0, 0), cv);
    cv = wmma_bf16(ak1, load_b_nmajor(lwv, DD, n0, 32), cv);
    store_c_bf16(Qo, DD, r0, n0, cq, 0.125f);  // 1/sqrt(D)
    store_c_bf16(Ko, DD, r0, n0, ck, 1.0f);
#pragma unroll
    for (int r = 0; r < 8; ++r)  // transposed V store: Vt[d][row]
      Vo[(size_t)(n0 + n) * LL + r0 + mb + r] = (__bf16)cv[r];
  }
}

// ---------------- Kernel 2: QE = Q @ Ek^T  (L x 80 per head, f32) ----------------
__global__ __launch_bounds__(256) void k_qe(const __bf16* __restrict__ Qbf,
                                            const float* __restrict__ Ek,
                                            float* __restrict__ QE) {
  int bh = blockIdx.x;
  int r0 = blockIdx.y * 128 + (int)(threadIdx.x >> 5) * 16;
  const __bf16* Qo = Qbf + (size_t)bh * LL * DD;
  float* QEo = QE + (size_t)bh * LL * QE_LD;
  v16bf a0 = load_a_bf16(Qo, DD, r0, 0);
  v16bf a1 = load_a_bf16(Qo, DD, r0, 32);
#pragma unroll
  for (int n0 = 0; n0 < QE_LD; n0 += 16) {
    v8f c = {};
    c = wmma_bf16(a0, load_b_f32_nmajor_g(Ek, DD, n0, 0, NPOS), c);
    c = wmma_bf16(a1, load_b_f32_nmajor_g(Ek, DD, n0, 32, NPOS), c);
    store_c_f32(QEo, QE_LD, r0, n0, c);
  }
}

// ---------------- Kernel 3: fused relative-position attention ----------------
// 128 threads = 4 waves; each wave owns one 16-row q tile of one (b,h).
// K panel is staged into LDS in double-buffered CHK-key chunks via the TDM.
// Two passes (exact softmax); rel logits gathered from QE; rel values via
// LDS bucket accumulation (ds_add_f32) + one small WMMA GEMM against Ev.
__global__ __launch_bounds__(128) void k_attn(const __bf16* __restrict__ Qbf,
                                              const __bf16* __restrict__ Kbf,
                                              const __bf16* __restrict__ Vt,
                                              const float* __restrict__ QE,
                                              const float* __restrict__ Ev,
                                              float* __restrict__ mh) {
  __shared__ __align__(16) float  ap[4 * 16 * AP_LD];    // 24 KB attn_pos per wave
  __shared__ __align__(16) __bf16 pst[4 * 16 * 32];      //  4 KB P staging per wave
  __shared__ __align__(16) __bf16 kstage[2][CHK * DD];   // 32 KB double-buffered K

  int bh = blockIdx.x;
  int b = bh / HH, h = bh % HH;
  int w = threadIdx.x >> 5;
  int lane = threadIdx.x & 31;
  int n = lane & 15, mb = (lane & 16) ? 8 : 0;
  int q0 = blockIdx.y * 64 + w * 16;

  const __bf16* Qo = Qbf + (size_t)bh * LL * DD;
  const __bf16* Ko = Kbf + (size_t)bh * LL * DD;
  const __bf16* Vo = Vt + (size_t)bh * DD * LL;
  const float* QEo = QE + (size_t)bh * LL * QE_LD;
  float* apw = ap + w * 16 * AP_LD;
  __bf16* pw = pst + w * 16 * 32;

  v16bf qa0 = load_a_bf16(Qo, DD, q0, 0);
  v16bf qa1 = load_a_bf16(Qo, DD, q0, 32);

  // ---- pass 1: exact row max / sum (online on scalars only) ----
  float mrow[8], srow[8];
#pragma unroll
  for (int r = 0; r < 8; ++r) { mrow[r] = -3.0e38f; srow[r] = 0.0f; }

  stage_k(Ko, kstage[0], 0);
  for (int ch = 0; ch < NCH; ++ch) {
    bool more = (ch + 1 < NCH);
    if (more) stage_k(Ko, kstage[(ch + 1) & 1], (ch + 1) * CHK);
    wait_k(more);
    const __bf16* kc = kstage[ch & 1];
    for (int kt = 0; kt < CHK; kt += 16) {
      int k0 = ch * CHK + kt;
      v8f s = {};
      s = wmma_bf16(qa0, load_b_nmajor(kc, DD, kt, 0), s);
      s = wmma_bf16(qa1, load_b_nmajor(kc, DD, kt, 32), s);
#pragma unroll
      for (int r = 0; r < 8; ++r) {
        int qrow = q0 + mb + r;
        int d = k0 + n - qrow;
        d = min(max(d, -MAXD), MAXD);
        float sv = s[r] + QEo[(size_t)qrow * QE_LD + d + MAXD];
        float tm = sv;
#pragma unroll
        for (int off = 1; off < 16; off <<= 1) tm = fmaxf(tm, __shfl_xor(tm, off, 32));
        float mnew = fmaxf(mrow[r], tm);
        float ts = __expf(sv - mnew);
#pragma unroll
        for (int off = 1; off < 16; off <<= 1) ts += __shfl_xor(ts, off, 32);
        srow[r] = srow[r] * __expf(mrow[r] - mnew) + ts;
        mrow[r] = mnew;
      }
    }
    __syncthreads();  // all waves done with this chunk before it is overwritten
  }
  float inv[8];
#pragma unroll
  for (int r = 0; r < 8; ++r) inv[r] = 1.0f / srow[r];

  for (int i = lane; i < 16 * AP_LD; i += 32) apw[i] = 0.0f;
  __syncthreads();

  // ---- pass 2: final P, O += P@V, bucket-accumulate P into attn_pos ----
  v8f o0 = {}, o1 = {}, o2 = {}, o3 = {};
  stage_k(Ko, kstage[0], 0);
  for (int ch = 0; ch < NCH; ++ch) {
    bool more = (ch + 1 < NCH);
    if (more) stage_k(Ko, kstage[(ch + 1) & 1], (ch + 1) * CHK);
    wait_k(more);
    const __bf16* kc = kstage[ch & 1];
    for (int kt = 0; kt < CHK; kt += 32) {
      int k0 = ch * CHK + kt;
      if (k0 + 32 < LL) __builtin_prefetch(Vo + (size_t)k0 + 32, 0, 0);
#pragma unroll
      for (int half = 0; half < 2; ++half) {
        int kb = kt + half * 16;
        v8f s = {};
        s = wmma_bf16(qa0, load_b_nmajor(kc, DD, kb, 0), s);
        s = wmma_bf16(qa1, load_b_nmajor(kc, DD, kb, 32), s);
#pragma unroll
        for (int r = 0; r < 8; ++r) {
          int qrow = q0 + mb + r;
          int d = k0 + half * 16 + n - qrow;
          d = min(max(d, -MAXD), MAXD);
          float sv = s[r] + QEo[(size_t)qrow * QE_LD + d + MAXD];
          float p = __expf(sv - mrow[r]) * inv[r];
          atomicAdd(&apw[(mb + r) * AP_LD + d + MAXD], p);     // ds_add_f32
          pw[(mb + r) * 32 + half * 16 + n] = (__bf16)p;       // D-layout -> staging
        }
      }
      __syncthreads();
      v16bf pa = load_a_bf16(pw, 32, 0, 0);  // staging -> A-layout
      o0 = wmma_bf16(pa, load_b_nmajor(Vo, LL, 0, k0), o0);
      o1 = wmma_bf16(pa, load_b_nmajor(Vo, LL, 16, k0), o1);
      o2 = wmma_bf16(pa, load_b_nmajor(Vo, LL, 32, k0), o2);
      o3 = wmma_bf16(pa, load_b_nmajor(Vo, LL, 48, k0), o3);
      __syncthreads();
    }
    __syncthreads();  // chunk fully consumed before the buffer is re-staged
  }

  // ---- rel-value term: O += attn_pos(16x96) @ Ev_padded(96x64) ----
#pragma unroll
  for (int kk = 0; kk < AP_LD; kk += 32) {
    v16bf pa = load_a_f32(apw, AP_LD, 0, kk);
    o0 = wmma_bf16(pa, load_b_f32_kmajor_g(Ev, DD, kk, 0, NPOS), o0);
    o1 = wmma_bf16(pa, load_b_f32_kmajor_g(Ev, DD, kk, 16, NPOS), o1);
    o2 = wmma_bf16(pa, load_b_f32_kmajor_g(Ev, DD, kk, 32, NPOS), o2);
    o3 = wmma_bf16(pa, load_b_f32_kmajor_g(Ev, DD, kk, 48, NPOS), o3);
  }

  // mh layout (B, L, H, D) f32
#pragma unroll
  for (int r = 0; r < 8; ++r) {
    int row = q0 + mb + r;
    size_t base = (((size_t)b * LL + row) * HH + h) * DD;
    mh[base + 0 + n]  = o0[r];
    mh[base + 16 + n] = o1[r];
    mh[base + 32 + n] = o2[r];
    mh[base + 48 + n] = o3[r];
  }
}

// ---------------- Kernel 4: output projection (2048 x 512) @ (512 x 64) + bias ----
__global__ __launch_bounds__(256) void k_out(const float* __restrict__ mh,
                                             const float* __restrict__ Wp,
                                             const float* __restrict__ bias,
                                             float* __restrict__ out) {
  int w = threadIdx.x >> 5;
  int lane = threadIdx.x & 31;
  int n = lane & 15, mb = (lane & 16) ? 8 : 0;
  const int KD = HH * DD;  // 512; Wp flat is already k-major: Wp[(h*D+i)*D + o]
  float b0 = bias[0 + n], b1 = bias[16 + n], b2 = bias[32 + n], b3 = bias[48 + n];
#pragma unroll
  for (int t = 0; t < 2; ++t) {
    int r0 = blockIdx.x * 256 + (w * 2 + t) * 16;
    v8f c0 = {}, c1 = {}, c2 = {}, c3 = {};
    for (int kk = 0; kk < KD; kk += 32) {
      v16bf a = load_a_f32(mh, KD, r0, kk);
      c0 = wmma_bf16(a, load_b_f32_kmajor_g(Wp, DD, kk, 0, KD), c0);
      c1 = wmma_bf16(a, load_b_f32_kmajor_g(Wp, DD, kk, 16, KD), c1);
      c2 = wmma_bf16(a, load_b_f32_kmajor_g(Wp, DD, kk, 32, KD), c2);
      c3 = wmma_bf16(a, load_b_f32_kmajor_g(Wp, DD, kk, 48, KD), c3);
    }
#pragma unroll
    for (int r = 0; r < 8; ++r) {
      int row = r0 + mb + r;
      out[(size_t)row * DD + 0 + n]  = c0[r] + b0;
      out[(size_t)row * DD + 16 + n] = c1[r] + b1;
      out[(size_t)row * DD + 32 + n] = c2[r] + b2;
      out[(size_t)row * DD + 48 + n] = c3[r] + b3;
    }
  }
}

extern "C" void kernel_launch(void* const* d_in, const int* in_sizes, int n_in,
                              void* d_out, int out_size, void* d_ws, size_t ws_size,
                              hipStream_t stream) {
  const float* q    = (const float*)d_in[0];
  const float* kv   = (const float*)d_in[1];
  const float* Wq   = (const float*)d_in[2];
  const float* Wk   = (const float*)d_in[3];
  const float* Wv   = (const float*)d_in[4];
  const float* Wp   = (const float*)d_in[5];
  const float* Ek   = (const float*)d_in[6];
  const float* Ev   = (const float*)d_in[7];
  const float* bias = (const float*)d_in[8];

  char* ws = (char*)d_ws;
  size_t off = 0;
  auto alloc = [&](size_t bytes) -> void* {
    void* p = ws + off;
    off += (bytes + 255) & ~(size_t)255;
    return p;
  };
  __bf16* Qbf = (__bf16*)alloc((size_t)BB * HH * LL * DD * 2);      // 2 MB
  __bf16* Kbf = (__bf16*)alloc((size_t)BB * HH * LL * DD * 2);      // 2 MB
  __bf16* Vt  = (__bf16*)alloc((size_t)BB * HH * DD * LL * 2);      // 2 MB
  float*  QE  = (float*)alloc((size_t)BB * HH * LL * QE_LD * 4);    // 5.25 MB
  float*  mh  = (float*)alloc((size_t)BB * LL * HH * DD * 4);       // 4 MB
  (void)ws_size;  // total ~15.7 MB

  k_proj<<<dim3(BB * HH, LL / 128), 256, 0, stream>>>(q, kv, Wq, Wk, Wv, Qbf, Kbf, Vt);
  k_qe  <<<dim3(BB * HH, LL / 128), 256, 0, stream>>>(Qbf, Ek, QE);
  k_attn<<<dim3(BB * HH, LL / 64), 128, 0, stream>>>(Qbf, Kbf, Vt, QE, Ev, mh);
  k_out <<<dim3((BB * LL) / 256), 256, 0, stream>>>(mh, Wp, bias, (float*)d_out);
}
